// GAT_55490977465228
// MI455X (gfx1250) — matrix operand order repaired and checked by reading, not compiled, not run on hardware
//
#include <hip/hip_runtime.h>

typedef __attribute__((ext_vector_type(16))) __bf16 v16bf;
typedef __attribute__((ext_vector_type(8)))  float  v8f;

// ---------- weight repack: Wt[c*128 + k] = bf16(W[k*K + c]) ----------
__global__ void k_repack_w(const float* __restrict__ W, __bf16* __restrict__ Wt, int K) {
  int idx = blockIdx.x * blockDim.x + threadIdx.x;   // over K*128 elements
  int c = idx >> 7, k = idx & 127;
  Wt[idx] = (__bf16)W[k * K + c];
}

// ---------- xl[N,K] = h[N,128] @ W[128,K] with v_wmma_f32_16x16x32_bf16 ----------
__global__ void k_gemm_wmma(const float* __restrict__ A, const __bf16* __restrict__ Wt,
                            float* __restrict__ C, int N, int K) {
  const int DIN = 128;
  int wave = threadIdx.x >> 5;
  int lane = threadIdx.x & 31;
  int half = lane >> 4;
  int l16  = lane & 15;
  int row  = blockIdx.x * 16 + l16;
  if (row >= N) row = N - 1;            // clamp loads; WMMA needs full EXEC
  int col  = wave * 16 + l16;

  v8f acc = {};
#pragma unroll
  for (int k0 = 0; k0 < DIN; k0 += 32) {
    // A fragment (contiguous k order, same as B): half 0 -> K {0..15}, half 1 -> K {16..31}
    const float* ap = A + (size_t)row * DIN + k0 + half * 16;
    v16bf a, b;
#pragma unroll
    for (int i = 0; i < 8; ++i) a[i]     = (__bf16)ap[i];
#pragma unroll
    for (int i = 0; i < 8; ++i) a[8 + i] = (__bf16)ap[8 + i];
    b = *(const v16bf*)(Wt + (size_t)col * DIN + k0 + half * 16);
    acc = __builtin_amdgcn_wmma_f32_16x16x32_bf16(false, a, false, b, (short)0, acc,
                                                  false, false);
  }
  int rbase = blockIdx.x * 16 + half * 8;
#pragma unroll
  for (int v = 0; v < 8; ++v) {
    int r = rbase + v;
    if (r < N) C[(size_t)r * K + wave * 16 + l16] = acc[v];
  }
}

// ---------- per-node attention logits: dot(xl[n,h,:], a_s/a_d[h]) ----------
__global__ void k_node_alpha(const float* __restrict__ xl, const float* __restrict__ a_s,
                             const float* __restrict__ a_d, float* __restrict__ as_n,
                             float* __restrict__ ad_n, int N, int K, int D) {
  int idx = blockIdx.x * blockDim.x + threadIdx.x;
  if (idx >= 2 * N) return;
  int n = idx >> 1, h = idx & 1;
  const float* xp = xl + (size_t)n * K + h * D;
  const float* sp = a_s + h * D;
  const float* dp = a_d + h * D;
  float ss = 0.f, dd = 0.f;
  for (int i = 0; i < D; ++i) { float v = xp[i]; ss += v * sp[i]; dd += v * dp[i]; }
  as_n[idx] = ss; ad_n[idx] = dd;
}

// ================= CSR build (once per launch; graph static across layers) ==========
__global__ void k_deg(const int* __restrict__ dst, int* __restrict__ deg, int E) {
  int e = blockIdx.x * blockDim.x + threadIdx.x;
  if (e < E) atomicAdd(&deg[dst[e]], 1);
}

// blockwise exclusive scan of deg -> rs, block totals -> bsum
__global__ void k_scan_block(const int* __restrict__ deg, int* __restrict__ rs,
                             int* __restrict__ bsum, int N) {
  __shared__ int sm[256];
  int i = blockIdx.x * 256 + threadIdx.x;
  int v = (i < N) ? deg[i] : 0;
  int x = v;
  sm[threadIdx.x] = x;
  __syncthreads();
#pragma unroll
  for (int o = 1; o < 256; o <<= 1) {
    int t = (threadIdx.x >= o) ? sm[threadIdx.x - o] : 0;
    __syncthreads();
    x += t;
    sm[threadIdx.x] = x;
    __syncthreads();
  }
  if (i < N) rs[i] = x - v;                       // exclusive
  if (threadIdx.x == 255) bsum[blockIdx.x] = x;   // block total
}

__global__ void k_scan_bsums(int* __restrict__ bsum, int nb) {
  if (blockIdx.x == 0 && threadIdx.x == 0) {
    int run = 0;
    for (int j = 0; j < nb; ++j) { int t = bsum[j]; bsum[j] = run; run += t; }
  }
}

__global__ void k_scan_add(int* __restrict__ rs, const int* __restrict__ bsum, int N) {
  int i = blockIdx.x * 256 + threadIdx.x;
  if (i < N) rs[i] += bsum[i >> 8];
}

__global__ void k_fill(const int* __restrict__ src, const int* __restrict__ dst,
                       int* __restrict__ cursor, int* __restrict__ csr_src, int E) {
  int e = blockIdx.x * blockDim.x + threadIdx.x;
  if (e >= E) return;
  int pos = atomicAdd(&cursor[dst[e]], 1);
  csr_src[pos] = src[e];
}

// ====== fused, atomic-free GAT aggregation: one wave32 per destination node =========
// online softmax over in-edges + weighted feature sum + bias/ReLU epilogue.
// Lane owns KPL contiguous features; K = 32*KPL; head h = lane*KPL / D.
template <int KPL>
__global__ void k_aggregate(const int* __restrict__ row_start, const int* __restrict__ csr_src,
                            const float* __restrict__ as_n, const float* __restrict__ ad_n,
                            const float* __restrict__ xl, const float* __restrict__ bias,
                            float* __restrict__ out, int N, int E, int D, int do_relu) {
  const int K = 32 * KPL;
  int lane = threadIdx.x & 31;
  int d = blockIdx.x * (blockDim.x >> 5) + (threadIdx.x >> 5);
  if (d >= N) return;                               // uniform per wave
  int h = (lane * KPL) / D;                         // 0 or 1
  float adst = ad_n[d * 2 + h];
  int beg = row_start[d];
  int end = (d == N - 1) ? E : row_start[d + 1];

  float m = -__builtin_inff();
  float den = 0.f;
  float acc[KPL];
#pragma unroll
  for (int i = 0; i < KPL; ++i) acc[i] = 0.f;

  for (int j = beg; j < end; ++j) {
    int s = csr_src[j];                             // same addr all lanes -> L0 broadcast
    float e = as_n[s * 2 + h] + adst;
    e = e > 0.f ? e : 0.2f * e;                     // leaky_relu 0.2
    float mn = fmaxf(m, e);
    float scale = __expf(m - mn);                   // exp(-inf)=0 on first edge
    float p = __expf(e - mn);
    den = den * scale + p;
    const float* xp = xl + (size_t)s * K + lane * KPL;
    if constexpr (KPL == 4) {
      float4 v = *(const float4*)xp;                // 32 lanes -> one 512B row
      acc[0] = acc[0] * scale + p * v.x;
      acc[1] = acc[1] * scale + p * v.y;
      acc[2] = acc[2] * scale + p * v.z;
      acc[3] = acc[3] * scale + p * v.w;
    } else {
      float2 v = *(const float2*)xp;
      acc[0] = acc[0] * scale + p * v.x;
      acc[1] = acc[1] * scale + p * v.y;
    }
    m = mn;
  }

  float inv = 1.f / (den + 1e-16f);                 // isolated node -> 0 (matches ref)
  float* op = out + (size_t)d * K + lane * KPL;
#pragma unroll
  for (int i = 0; i < KPL; ++i) {
    float v = acc[i] * inv;
    if (bias) {
      v += bias[lane * KPL + i];
      if (do_relu) v = v > 0.f ? v : 0.f;
    }
    op[i] = v;
  }
}

// ---------- final layer: head-mean + bias + log_softmax (one wave32 per node) ----------
__global__ void k_final(const float* __restrict__ agg, const float* __restrict__ b,
                        float* __restrict__ out, int N) {
  int lane = threadIdx.x & 31;
  int n = blockIdx.x * 8 + (threadIdx.x >> 5);
  if (n >= N) return;                               // uniform per wave
  float v = 0.5f * (agg[(size_t)n * 64 + lane] + agg[(size_t)n * 64 + 32 + lane]) + b[lane];
  float mx = v;
#pragma unroll
  for (int o = 16; o > 0; o >>= 1) mx = fmaxf(mx, __shfl_xor(mx, o, 32));
  float ex = __expf(v - mx);
  float sum = ex;
#pragma unroll
  for (int o = 16; o > 0; o >>= 1) sum += __shfl_xor(sum, o, 32);
  out[(size_t)n * 32 + lane] = (v - mx) - __logf(sum);
}

// ======================================================================
extern "C" void kernel_launch(void* const* d_in, const int* in_sizes, int n_in,
                              void* d_out, int out_size, void* d_ws, size_t ws_size,
                              hipStream_t stream) {
  const float* x    = (const float*)d_in[0];
  const int*   eidx = (const int*)d_in[1];
  const int N = in_sizes[0] / 128;
  const int E = in_sizes[1] / 2;
  const int* src = eidx;
  const int* dst = eidx + E;
  const int nb = (N + 255) / 256;

  size_t off = 0;
  auto carve = [&](size_t bytes) -> void* {
    void* p = (char*)d_ws + off;
    off += (bytes + 255) & ~(size_t)255;
    return p;
  };
  float*  xl      = (float*)carve((size_t)N * 128 * 4);
  float*  hbuf    = (float*)carve((size_t)N * 128 * 4);
  float*  as_n    = (float*)carve((size_t)N * 2 * 4);
  float*  ad_n    = (float*)carve((size_t)N * 2 * 4);
  int*    deg     = (int*)  carve((size_t)N * 4);
  int*    rs      = (int*)  carve((size_t)N * 4);   // row_start (exclusive prefix sum)
  int*    cursor  = (int*)  carve((size_t)N * 4);
  int*    bsum    = (int*)  carve((size_t)nb * 4);
  int*    csr_src = (int*)  carve((size_t)E * 4);
  __bf16* Wt      = (__bf16*)carve(128 * 128 * 2);

  // ---- build destination-sorted CSR once (graph shared by all 4 layers) ----
  hipMemsetAsync(deg, 0, (size_t)N * 4, stream);
  k_deg       <<<(E + 255) / 256, 256, 0, stream>>>(dst, deg, E);
  k_scan_block<<<nb, 256, 0, stream>>>(deg, rs, bsum, N);
  k_scan_bsums<<<1, 32, 0, stream>>>(bsum, nb);
  k_scan_add  <<<nb, 256, 0, stream>>>(rs, bsum, N);
  hipMemcpyAsync(cursor, rs, (size_t)N * 4, hipMemcpyDeviceToDevice, stream);
  k_fill      <<<(E + 255) / 256, 256, 0, stream>>>(src, dst, cursor, csr_src, E);

  struct Layer { const float *W, *as, *ad, *b; int K, D; };
  Layer L[4] = {
    {(const float*)d_in[2],  (const float*)d_in[3],  (const float*)d_in[4],  (const float*)d_in[5],  128, 64},
    {(const float*)d_in[6],  (const float*)d_in[7],  (const float*)d_in[8],  (const float*)d_in[9],  128, 64},
    {(const float*)d_in[10], (const float*)d_in[11], (const float*)d_in[12], (const float*)d_in[13], 128, 64},
    {(const float*)d_in[14], (const float*)d_in[15], (const float*)d_in[16], (const float*)d_in[17],  64, 32},
  };

  const float* h_cur = x;   // gemm reads h_cur -> xl; aggregate writes hbuf (stream-ordered)
  for (int li = 0; li < 4; ++li) {
    const int K = L[li].K, D = L[li].D;

    k_repack_w  <<<(128 * K + 255) / 256, 256, 0, stream>>>(L[li].W, Wt, K);
    k_gemm_wmma <<<(N + 15) / 16, (K / 16) * 32, 0, stream>>>(h_cur, Wt, xl, N, K);
    k_node_alpha<<<(2 * N + 255) / 256, 256, 0, stream>>>(xl, L[li].as, L[li].ad,
                                                          as_n, ad_n, N, K, D);
    if (li < 3) {
      k_aggregate<4><<<(N + 7) / 8, 256, 0, stream>>>(rs, csr_src, as_n, ad_n, xl,
                                                      L[li].b, hbuf, N, E, D, 1);
      h_cur = hbuf;
    } else {
      k_aggregate<2><<<(N + 7) / 8, 256, 0, stream>>>(rs, csr_src, as_n, ad_n, xl,
                                                      nullptr, hbuf, N, E, D, 0);
      k_final<<<(N + 7) / 8, 256, 0, stream>>>(hbuf, L[li].b, (float*)d_out, N);
    }
  }
}
